// GraphLayer_29042568855655
// MI455X (gfx1250) — matrix operand order repaired and checked by reading, not compiled, not run on hardware
//
#include <hip/hip_runtime.h>

// ---------------------------------------------------------------------------
// Problem constants (from reference setup_inputs)
// ---------------------------------------------------------------------------
constexpr int Bn = 8;      // batch
constexpr int Cn = 128;    // in channels
constexpr int Nn = 4096;   // points
constexpr int On = 256;    // out channels
constexpr int Kn = 16;     // knn k
constexpr int PAD = 8;     // LDS panel row padding (halves) -> bank-conflict free
constexpr float BN_EPS = 1e-5f;

typedef _Float16 half_t;
typedef __attribute__((ext_vector_type(16))) _Float16 v16h;
typedef __attribute__((ext_vector_type(8)))  _Float16 v8h;
typedef __attribute__((ext_vector_type(8)))  float    v8f;
typedef __attribute__((ext_vector_type(4)))  int      v4i;

// ---------------------------------------------------------------------------
// CDNA5 async global->LDS staging (ASYNCcnt-tracked DMA path).
// Builtin signature (probe-confirmed):
//   void (v4i addrspace(1)* src, v4i addrspace(3)* dst, imm int off, imm cpol)
// ---------------------------------------------------------------------------
__device__ __forceinline__ void async_copy16(void* ldsdst, const void* gsrc) {
#if __has_builtin(__builtin_amdgcn_global_load_async_to_lds_b128)
  typedef __attribute__((address_space(1))) v4i gv4i_t;
  typedef __attribute__((address_space(3))) v4i lv4i_t;
  __builtin_amdgcn_global_load_async_to_lds_b128((gv4i_t*)gsrc, (lv4i_t*)ldsdst,
                                                 0, 0);
#else
  asm volatile("global_load_async_to_lds_b128 %0, %1, off"
               :
               : "v"((__attribute__((address_space(3))) void*)ldsdst),
                 "v"(gsrc)
               : "memory");
#endif
}

__device__ __forceinline__ void wait_async0() {
#if __has_builtin(__builtin_amdgcn_s_wait_asynccnt)
  __builtin_amdgcn_s_wait_asynccnt(0);
#else
  asm volatile("s_wait_asynccnt 0x0" ::: "memory");
#endif
}

// ---------------------------------------------------------------------------
// K1: x (B,C,N) f32 -> xt (B,N,C) f16   +   xx[b][n] = sum_c x^2
// ---------------------------------------------------------------------------
__global__ __launch_bounds__(256) void k_prep(const float* __restrict__ x,
                                              half_t* __restrict__ xt,
                                              float* __restrict__ xx) {
  int gid = blockIdx.x * blockDim.x + threadIdx.x;   // over B*N
  if (gid >= Bn * Nn) return;
  int b = gid / Nn, n = gid % Nn;
  const float* xp = x + (size_t)b * Cn * Nn + n;
  half_t* op = xt + (size_t)gid * Cn;
  float s = 0.f;
#pragma unroll 4
  for (int c = 0; c < Cn; ++c) {
    float v = xp[(size_t)c * Nn];
    s += v * v;
    op[c] = (half_t)v;
  }
  xx[gid] = s;
}

// ---------------------------------------------------------------------------
// K1b: conv_w (O,C) f32 -> f16
// ---------------------------------------------------------------------------
__global__ __launch_bounds__(256) void k_wconv(const float* __restrict__ w,
                                               half_t* __restrict__ wh) {
  int i = blockIdx.x * blockDim.x + threadIdx.x;
  if (i < On * Cn) wh[i] = (half_t)w[i];
}

// ---------------------------------------------------------------------------
// Fragment loaders (contiguous k order used here, k = 16h + s; A and B match).
//   A 16x32 (MxK): lanes 0-15 -> M=lane,   K = kb+{0..15}
//                  lanes16-31 -> M=lane-16,K = kb+{16..31}
//   B 32x16 (KxN): lanes 0-15 -> N=lane,   K = kb+{0..15}
//                  lanes16-31 -> N=lane-16,K = kb+{16..31}
// ---------------------------------------------------------------------------
__device__ __forceinline__ v16h load_afrag(const half_t* rowp, int kb, int hi) {
  int k0 = kb + hi * 16;
  v8h lo = *(const v8h*)(rowp + k0);
  v8h hh = *(const v8h*)(rowp + k0 + 8);
  v16h a;
#pragma unroll
  for (int i = 0; i < 8; ++i) { a[i] = lo[i]; a[8 + i] = hh[i]; }
  return a;
}
// B fragment from 32B-aligned global rows: single v16h deref.
__device__ __forceinline__ v16h load_bfrag(const half_t* colp, int kb, int hi) {
  return *(const v16h*)(colp + kb + hi * 16);
}
// B fragment from padded LDS rows (16B-aligned only): two v8h derefs.
__device__ __forceinline__ v16h load_bfrag_lds(const half_t* colp, int kb, int hi) {
  int k0 = kb + hi * 16;
  v8h lo = *(const v8h*)(colp + k0);
  v8h hh = *(const v8h*)(colp + k0 + 8);
  v16h b;
#pragma unroll
  for (int i = 0; i < 8; ++i) { b[i] = lo[i]; b[8 + i] = hh[i]; }
  return b;
}

// ---------------------------------------------------------------------------
// K2: fused Gram-GEMM + pairwise-distance + running top-16 per row.
//
// Block = 256 threads (8 waves). Wave w owns row tile r0 = blk*128 + w*16 and
// keeps its A fragments (16 rows x 128 K, f16) in VGPRs for the whole kernel.
// The shared 16-column B panel is staged into LDS with
// global_load_async_to_lds_b128, double-buffered: the ASYNCcnt DMA of panel
// m+1 overlaps WMMA + top-k merge of panel m. Panel rows are padded by 16 B
// (row stride 272 B = 4 banks mod 64) so the 16 half-wave lanes' ds_load_b128
// accesses tile all 64 LDS banks exactly once -> conflict-free.
// The (N,N) distance matrix is never materialized.
// ---------------------------------------------------------------------------
__global__ __launch_bounds__(256) void k_gram_topk(const half_t* __restrict__ xt,
                                                   const float* __restrict__ xx,
                                                   int* __restrict__ idxout) {
  const int t    = threadIdx.x;        // 0..255
  const int lane = t & 31;
  const int w    = t >> 5;             // wave id 0..7
  const int b    = blockIdx.y;
  const int r0   = blockIdx.x * 128 + w * 16;
  const int m15  = lane & 15;
  const int hi   = lane >> 4;          // 0 or 1
  const int rbase = hi * 8;

  const half_t* xtb = xt + (size_t)b * Nn * Cn;
  const float*  xxb = xx + (size_t)b * Nn;

  __shared__ half_t bpanel[2][16][Cn + PAD];  // double-buffered, padded panel
  __shared__ float  mtile[8][16][17];         // per-wave pdist tile for merge

  // A fragments for this wave's row tile, all K chunks, in VGPRs.
  const half_t* rowp = xtb + (size_t)(r0 + m15) * Cn;
  v16h afrag[4];
#pragma unroll
  for (int kk = 0; kk < 4; ++kk) afrag[kk] = load_afrag(rowp, kk * 32, hi);

  // Row norms needed by this lane's 8 output rows, in registers.
  float xxr[8];
#pragma unroll
  for (int v = 0; v < 8; ++v) xxr[v] = xxb[r0 + rbase + v];

  // Per-row top-16 (sorted descending), register resident (lanes 0-15).
  float vals[16]; int idxs[16];
#pragma unroll
  for (int i = 0; i < 16; ++i) { vals[i] = -3.4e38f; idxs[i] = 0; }

  // Panel staging: thread t copies 16B: row = t>>4 (gram column), seg = t&15.
  const int prow = t >> 4, pseg = t & 15;
  const half_t* gpan0 = xtb + (size_t)prow * Cn + pseg * 8;

  int buf = 0;
  async_copy16(&bpanel[0][prow][pseg * 8], gpan0);   // panel for m0 = 0

  for (int m0 = 0; m0 < Nn; m0 += 16) {
    wait_async0();
    __syncthreads();                   // panel[buf] visible to all waves

    if (m0 + 16 < Nn)                  // kick DMA for the next panel
      async_copy16(&bpanel[buf ^ 1][prow][pseg * 8],
                   gpan0 + (size_t)(m0 + 16) * Cn);

    const half_t* colp = &bpanel[buf][m15][0];
    v8f acc = {};
#pragma unroll
    for (int kk = 0; kk < 4; ++kk) {
      v16h bfrag = load_bfrag_lds(colp, kk * 32, hi);   // conflict-free ds_load
      acc = __builtin_amdgcn_wmma_f32_16x16x32_f16(false, afrag[kk],
                                                   false, bfrag,
                                                   (short)0, acc,
                                                   false, false);
    }

    // pdist = 2*inner - xx[row] - xx[col]; stage tile for the per-row merge.
    float xxc = xxb[m0 + m15];
#pragma unroll
    for (int v = 0; v < 8; ++v)
      mtile[w][rbase + v][m15] = 2.0f * acc[v] - xxr[v] - xxc;
    __syncthreads();

    if (lane < 16) {
#pragma unroll
      for (int j = 0; j < 16; ++j) {
        float cv = mtile[w][lane][j];
        if (cv > vals[15]) {           // beats current min -> insert
          int ci = m0 + j;
#pragma unroll
          for (int s = 0; s < 16; ++s) {   // unrolled compare-swap insertion
            bool sw = cv > vals[s];
            float tv = vals[s]; int ti = idxs[s];
            vals[s] = sw ? cv : vals[s];
            idxs[s] = sw ? ci : idxs[s];
            cv = sw ? tv : cv;
            ci = sw ? ti : ci;
          }
        }
      }
    }
    __syncthreads();                   // reads of panel[buf] & mtile done
    buf ^= 1;
  }

  if (lane < 16) {
    int* op = idxout + ((size_t)b * Nn + r0 + lane) * Kn;
#pragma unroll
    for (int j = 0; j < 16; ++j) op[j] = idxs[j];
  }
}

// ---------------------------------------------------------------------------
// K3: gather 16 neighbors, max over k, emit pooled (B,N,C) f16.
// x per batch is 2 MB -> gather traffic is L2-resident.
// ---------------------------------------------------------------------------
__global__ __launch_bounds__(128) void k_pool(const float* __restrict__ x,
                                              const int* __restrict__ idx,
                                              half_t* __restrict__ pooled) {
  int bn = blockIdx.x;                 // b*N + n
  int b  = bn >> 12;                   // N = 4096
  int c  = threadIdx.x;                // 0..127
  __shared__ int nbr[16];
  if (c < 16) nbr[c] = idx[(size_t)bn * Kn + c];
  __syncthreads();
  const float* xb = x + (size_t)b * Cn * Nn + (size_t)c * Nn;
  float m = -3.4e38f;
#pragma unroll
  for (int j = 0; j < 16; ++j) m = fmaxf(m, xb[nbr[j]]);
  pooled[(size_t)bn * Cn + c] = (half_t)m;
}

// ---------------------------------------------------------------------------
// K4: 1x1 conv as WMMA GEMM: y(o,n) = wh(o,:) . pooled(n,:) + bias.
// One wave per block; 16-row (o) strip over 4 n-tiles, reusing A fragments.
// Writes pre-BN y into d_out (HBM-write bound; ~33 MB @ 23.3 TB/s).
// ---------------------------------------------------------------------------
__global__ __launch_bounds__(32) void k_conv(const half_t* __restrict__ wh,
                                             const half_t* __restrict__ pooled,
                                             const float* __restrict__ convb,
                                             float* __restrict__ y) {
  const int lane = threadIdx.x;
  const int m15  = lane & 15;
  const int hi   = lane >> 4;
  const int rbase = hi * 8;
  const int o0 = blockIdx.y * 16;
  const int b  = blockIdx.z;

  const half_t* rowp = wh + (size_t)(o0 + m15) * Cn;
  v16h afrag[4];
#pragma unroll
  for (int kk = 0; kk < 4; ++kk) afrag[kk] = load_afrag(rowp, kk * 32, hi);

  float bias[8];
#pragma unroll
  for (int v = 0; v < 8; ++v) bias[v] = convb[o0 + rbase + v];

  const half_t* pb = pooled + (size_t)b * Nn * Cn;
  float* yb = y + ((size_t)b * On + o0) * Nn;

  for (int tt = 0; tt < 4; ++tt) {
    int n0 = (blockIdx.x * 4 + tt) * 16;
    const half_t* colp = pb + (size_t)(n0 + m15) * Cn;
    v8f acc = {};
#pragma unroll
    for (int kk = 0; kk < 4; ++kk) {
      v16h bfrag = load_bfrag(colp, kk * 32, hi);
      acc = __builtin_amdgcn_wmma_f32_16x16x32_f16(false, afrag[kk],
                                                   false, bfrag,
                                                   (short)0, acc,
                                                   false, false);
    }
#pragma unroll
    for (int v = 0; v < 8; ++v)
      yb[(size_t)(rbase + v) * Nn + n0 + m15] = acc[v] + bias[v];
  }
}

// ---------------------------------------------------------------------------
// K5: per-channel batch statistics (mean, rsqrt(var+eps)) over (B,N).
// ---------------------------------------------------------------------------
__global__ __launch_bounds__(256) void k_stats(const float* __restrict__ y,
                                               float* __restrict__ stats) {
  int o = blockIdx.x;
  __shared__ float s1[256], s2[256];
  float a1 = 0.f, a2 = 0.f;
  for (int i = threadIdx.x; i < Bn * Nn; i += 256) {
    int b = i / Nn, n = i % Nn;
    float v = y[((size_t)b * On + o) * Nn + n];
    a1 += v; a2 += v * v;
  }
  s1[threadIdx.x] = a1; s2[threadIdx.x] = a2;
  __syncthreads();
  for (int s = 128; s > 0; s >>= 1) {
    if (threadIdx.x < s) {
      s1[threadIdx.x] += s1[threadIdx.x + s];
      s2[threadIdx.x] += s2[threadIdx.x + s];
    }
    __syncthreads();
  }
  if (threadIdx.x == 0) {
    const float inv = 1.0f / (float)(Bn * Nn);
    float mean = s1[0] * inv;
    float var  = s2[0] * inv - mean * mean;
    stats[o * 2 + 0] = mean;
    stats[o * 2 + 1] = rsqrtf(var + BN_EPS);
  }
}

// ---------------------------------------------------------------------------
// K6: in-place BatchNorm affine + ReLU on d_out.
// ---------------------------------------------------------------------------
__global__ __launch_bounds__(256) void k_bnrelu(float* __restrict__ y,
                                                const float* __restrict__ stats,
                                                const float* __restrict__ gamma,
                                                const float* __restrict__ beta) {
  size_t e = (size_t)blockIdx.x * blockDim.x + threadIdx.x;
  if (e >= (size_t)Bn * On * Nn) return;
  int o = (int)((e / Nn) % On);
  float v = (y[e] - stats[o * 2]) * stats[o * 2 + 1];
  v = gamma[o] * v + beta[o];
  y[e] = fmaxf(v, 0.f);
}

// ---------------------------------------------------------------------------
// kernel_launch
// ---------------------------------------------------------------------------
extern "C" void kernel_launch(void* const* d_in, const int* in_sizes, int n_in,
                              void* d_out, int out_size, void* d_ws, size_t ws_size,
                              hipStream_t stream) {
  (void)in_sizes; (void)n_in; (void)out_size; (void)ws_size;
  const float* x     = (const float*)d_in[0];
  const float* cw    = (const float*)d_in[1];
  const float* cb    = (const float*)d_in[2];
  const float* gamma = (const float*)d_in[3];
  const float* beta  = (const float*)d_in[4];
  float* y = (float*)d_out;

  // Workspace carve-up (all chunks 256B-aligned by construction). ~19 MB.
  char* p = (char*)d_ws;
  half_t* xt     = (half_t*)p; p += (size_t)Bn * Nn * Cn * sizeof(half_t);
  half_t* pooled = (half_t*)p; p += (size_t)Bn * Nn * Cn * sizeof(half_t);
  half_t* wh     = (half_t*)p; p += (size_t)On * Cn * sizeof(half_t);
  float*  xx     = (float*)p;  p += (size_t)Bn * Nn * sizeof(float);
  int*    idxws  = (int*)p;    p += (size_t)Bn * Nn * Kn * sizeof(int);
  float*  stats  = (float*)p;

  // 1) transpose/convert + norms
  k_prep<<<(Bn * Nn + 255) / 256, 256, 0, stream>>>(x, xt, xx);
  // 1b) weights -> f16
  k_wconv<<<(On * Cn + 255) / 256, 256, 0, stream>>>(cw, wh);
  // 2) fused Gram-WMMA + top-16 (async LDS panel staging, double buffered)
  k_gram_topk<<<dim3(Nn / 128, Bn), 256, 0, stream>>>(xt, xx, idxws);
  // 3) gather + max-pool over k
  k_pool<<<Bn * Nn, 128, 0, stream>>>(x, idxws, pooled);
  // 4) 1x1 conv (WMMA) + bias -> pre-BN y in d_out
  k_conv<<<dim3(Nn / 64, On / 16, Bn), 32, 0, stream>>>(wh, pooled, cb, y);
  // 5) batch stats
  k_stats<<<On, 256, 0, stream>>>(y, stats);
  // 6) BN + ReLU in place
  k_bnrelu<<<(int)(((size_t)Bn * On * Nn + 255) / 256), 256, 0, stream>>>(y, stats, gamma, beta);
}
